// Model_4020089389441
// MI455X (gfx1250) — matrix operand order repaired and checked by reading, not compiled, not run on hardware
//
#include <hip/hip_runtime.h>
#include <hip/hip_bf16.h>

#define N_NODES 50000
#define N_EDGES 1600000
#define HDIM    64
#define NCLS    10
#define SLOPE   0.01f

typedef __attribute__((ext_vector_type(16))) _Float16 v16h;
typedef __attribute__((ext_vector_type(8)))  float    v8f;

__device__ __forceinline__ float leaky(float v) { return v > 0.f ? v : SLOPE * v; }

// ---------------- utility kernels ----------------
__global__ void k_zero(float* __restrict__ p, int n) {
    int i = blockIdx.x * blockDim.x + threadIdx.x;
    if (i < n) p[i] = 0.f;
}

__global__ void k_fill1(float* __restrict__ p, int n) {
    int i = blockIdx.x * blockDim.x + threadIdx.x;
    if (i < n) p[i] = 1.f;              // self-loop contributes 1 to degree
}

__global__ void k_deg(const int* __restrict__ dst, float* __restrict__ deg, int e) {
    int i = blockIdx.x * blockDim.x + threadIdx.x;
    if (i < e) atomicAdd(&deg[dst[i]], 1.f);
}

__global__ void k_dinv(const float* __restrict__ deg, float* __restrict__ dinv, int n) {
    int i = blockIdx.x * blockDim.x + threadIdx.x;
    if (i < n) dinv[i] = rsqrtf(deg[i]);
}

// ---------------- layer 1 (in=1 -> out=10) ----------------
__global__ void k_l1(const float* __restrict__ x, const float* __restrict__ W1,
                     float* __restrict__ t, int n) {
    int idx = blockIdx.x * blockDim.x + threadIdx.x;
    if (idx >= n * NCLS) return;
    int i = idx / NCLS, j = idx - i * NCLS;
    t[idx] = x[i] * W1[j];
}

__global__ void k_scatter10(const float* __restrict__ t, const int* __restrict__ src,
                            const int* __restrict__ dst, const float* __restrict__ dinv,
                            float* __restrict__ agg, int e) {
    int i = blockIdx.x * blockDim.x + threadIdx.x;
    if (i >= e) return;
    int s = src[i], d = dst[i];
    float nrm = dinv[s] * dinv[d];
#pragma unroll
    for (int j = 0; j < NCLS; ++j)
        atomicAdd(&agg[d * NCLS + j], t[s * NCLS + j] * nrm);
}

// write h1 zero-padded to stride 16 so the WMMA A-fragment loads are branch-free
__global__ void k_finish10p(const float* __restrict__ agg, const float* __restrict__ t,
                            const float* __restrict__ dinv, const float* __restrict__ b,
                            float* __restrict__ h, int n) {
    int idx = blockIdx.x * blockDim.x + threadIdx.x;
    if (idx >= n * 16) return;
    int i = idx >> 4, j = idx & 15;
    float v = 0.f;
    if (j < NCLS) {
        float di = dinv[i];
        v = leaky(agg[i * NCLS + j] + t[i * NCLS + j] * di * di + b[j]);
    }
    h[idx] = v;
}

// ---------------- WMMA B-fragment pre-swizzle ----------------
// frag layout: [(ks*4 + ctile)*32 + lane] -> v16h; elem i: K=ks*32+8*(lane>=16)+(i&7)+16*(i>=8)
__global__ void k_wfrag(const float* __restrict__ W, _Float16* __restrict__ frag,
                        int rows, int ksteps) {
    int idx = blockIdx.x * blockDim.x + threadIdx.x;
    int total = ksteps * 4 * 32 * 16;
    if (idx >= total) return;
    int i    = idx & 15;
    int lane = (idx >> 4) & 31;
    int ct   = (idx >> 9) & 3;
    int ks   = idx >> 11;
    int K    = ks * 32 + ((lane & 16) ? 8 : 0) + (i & 7) + ((i & 8) ? 16 : 0);
    int col  = ct * 16 + (lane & 15);
    frag[idx] = (K < rows) ? (_Float16)W[K * HDIM + col] : (_Float16)0.f;
}

// ---------------- WMMA GEMM: C[N x 64] = A[N x KPAD(zero-padded)] @ W ----------------
// 128 threads = 4 waves; wave w owns column tile w; one 16x16 C tile per wave.
// KPAD is the (zero-padded) row stride of A; K>=KPAD contributions are compile-time zero.
__device__ __forceinline__ void load8h(v16h& a, int off, const float4* __restrict__ row, int k) {
    float4 p = row[k >> 2], q = row[(k >> 2) + 1];
    a[off + 0] = (_Float16)p.x; a[off + 1] = (_Float16)p.y;
    a[off + 2] = (_Float16)p.z; a[off + 3] = (_Float16)p.w;
    a[off + 4] = (_Float16)q.x; a[off + 5] = (_Float16)q.y;
    a[off + 6] = (_Float16)q.z; a[off + 7] = (_Float16)q.w;
}

template <int KSTEPS, int KPAD>
__global__ void k_gemm(const float* __restrict__ A, const _Float16* __restrict__ Bfrag,
                       float* __restrict__ C) {
    const int lane = threadIdx.x & 31;
    const int wave = threadIdx.x >> 5;          // column tile 0..3
    const int m0   = blockIdx.x * 16;           // N divisible by 16
    const int M    = m0 + (lane & 15);
    const int hs   = lane >> 4;                 // half-select per 16-bit A layout
    const float4* Arow = (const float4*)(A + (size_t)M * KPAD);
    const v16h* Bf = (const v16h*)Bfrag;
    v8f acc = {};
#pragma unroll
    for (int ks = 0; ks < KSTEPS; ++ks) {
        v16h a;
        if (ks * 32 < KPAD) {                   // compile-time
            load8h(a, 0, Arow, ks * 32 + hs * 8);
        } else {
#pragma unroll
            for (int i = 0; i < 8; ++i) a[i] = (_Float16)0.f;
        }
        if (ks * 32 + 16 < KPAD) {              // compile-time (false for KPAD=16)
            load8h(a, 8, Arow, ks * 32 + 16 + hs * 8);
        } else {
#pragma unroll
            for (int i = 8; i < 16; ++i) a[i] = (_Float16)0.f;
        }
        v16h b = Bf[(ks * 4 + wave) * 32 + lane];   // contiguous 32B fragment load
        acc = __builtin_amdgcn_wmma_f32_16x16x32_f16(
            /*neg_a=*/false, a, /*neg_b=*/false, b,
            /*c_mod=*/(short)0, acc, /*reuse_a=*/false, /*reuse_b=*/false);
    }
    const int col = wave * 16 + (lane & 15);
    const int rb  = m0 + hs * 8;                // C layout: VGPR v -> row rb+v
#pragma unroll
    for (int v = 0; v < 8; ++v)
        C[(size_t)(rb + v) * HDIM + col] = acc[v];
}

// ---------------- 64-wide aggregation ----------------
__global__ void k_scatter64(const float4* __restrict__ t, const int* __restrict__ src,
                            const int* __restrict__ dst, const float* __restrict__ dinv,
                            float* __restrict__ agg, int e) {
    int idx = blockIdx.x * blockDim.x + threadIdx.x;   // e*16 threads
    if (idx >= e * 16) return;
    int ed = idx >> 4, c = idx & 15;
    int s = src[ed], d = dst[ed];
    float nrm = dinv[s] * dinv[d];
    float4 v  = t[(size_t)s * 16 + c];
    float* ap = agg + (size_t)d * HDIM + c * 4;
    atomicAdd(ap + 0, v.x * nrm);
    atomicAdd(ap + 1, v.y * nrm);
    atomicAdd(ap + 2, v.z * nrm);
    atomicAdd(ap + 3, v.w * nrm);
}

__global__ void k_finish64(const float4* __restrict__ agg, const float4* __restrict__ t,
                           const float* __restrict__ dinv, const float4* __restrict__ b,
                           float4* __restrict__ h, int n) {
    int idx = blockIdx.x * blockDim.x + threadIdx.x;   // n*16
    if (idx >= n * 16) return;
    int i = idx >> 4, c = idx & 15;
    float d2 = dinv[i];
    d2 *= d2;
    float4 a = agg[idx], tv = t[idx], bv = b[c], r;
    r.x = leaky(a.x + tv.x * d2 + bv.x);
    r.y = leaky(a.y + tv.y * d2 + bv.y);
    r.z = leaky(a.z + tv.z * d2 + bv.z);
    r.w = leaky(a.w + tv.w * d2 + bv.w);
    h[idx] = r;
}

// ---------------- final FC reduction + softmax ----------------
// Two rows (80B, 16B-aligned) per thread -> five b128 loads for fcW.
__global__ void k_fc(const float* __restrict__ h, const float* __restrict__ fcW,
                     float* __restrict__ logits, int pairs) {
    __shared__ float sacc[NCLS];
    if (threadIdx.x < NCLS) sacc[threadIdx.x] = 0.f;
    __syncthreads();
    float acc[NCLS];
#pragma unroll
    for (int c = 0; c < NCLS; ++c) acc[c] = 0.f;
    const int stride = gridDim.x * blockDim.x;
    for (int p = blockIdx.x * blockDim.x + threadIdx.x; p < pairs; p += stride) {
        __builtin_prefetch(fcW + ((size_t)p + stride) * 2 * NCLS, 0, 0);
        float2 a = ((const float2*)h)[p];
        const float4* w = (const float4*)(fcW + (size_t)p * 2 * NCLS);
        float4 w0 = w[0], w1 = w[1], w2 = w[2], w3 = w[3], w4 = w[4];
        acc[0] += a.x * w0.x; acc[1] += a.x * w0.y; acc[2] += a.x * w0.z; acc[3] += a.x * w0.w;
        acc[4] += a.x * w1.x; acc[5] += a.x * w1.y; acc[6] += a.x * w1.z; acc[7] += a.x * w1.w;
        acc[8] += a.x * w2.x; acc[9] += a.x * w2.y;
        acc[0] += a.y * w2.z; acc[1] += a.y * w2.w;
        acc[2] += a.y * w3.x; acc[3] += a.y * w3.y; acc[4] += a.y * w3.z; acc[5] += a.y * w3.w;
        acc[6] += a.y * w4.x; acc[7] += a.y * w4.y; acc[8] += a.y * w4.z; acc[9] += a.y * w4.w;
    }
#pragma unroll
    for (int c = 0; c < NCLS; ++c) atomicAdd(&sacc[c], acc[c]);
    __syncthreads();
    if (threadIdx.x < NCLS) atomicAdd(&logits[threadIdx.x], sacc[threadIdx.x]);
}

__global__ void k_softmax(const float* __restrict__ logits, const float* __restrict__ fcb,
                          float* __restrict__ out) {
    if (threadIdx.x == 0 && blockIdx.x == 0) {
        float v[NCLS], mx = -1e30f;
        for (int c = 0; c < NCLS; ++c) { v[c] = logits[c] + fcb[c]; mx = fmaxf(mx, v[c]); }
        float s = 0.f;
        for (int c = 0; c < NCLS; ++c) { v[c] = expf(v[c] - mx); s += v[c]; }
        float inv = 1.f / s;
        for (int c = 0; c < NCLS; ++c) out[c] = v[c] * inv;
    }
}

// ---------------- host launch ----------------
extern "C" void kernel_launch(void* const* d_in, const int* in_sizes, int n_in,
                              void* d_out, int out_size, void* d_ws, size_t ws_size,
                              hipStream_t stream) {
    (void)in_sizes; (void)n_in; (void)out_size; (void)ws_size;
    const float* x   = (const float*)d_in[0];
    const int*   ei  = (const int*)d_in[1];
    const float* W1  = (const float*)d_in[2];
    const float* b1  = (const float*)d_in[3];
    const float* W2  = (const float*)d_in[4];
    const float* b2  = (const float*)d_in[5];
    const float* W3  = (const float*)d_in[6];
    const float* b3  = (const float*)d_in[7];
    const float* fcW = (const float*)d_in[8];
    const float* fcb = (const float*)d_in[9];
    float* out = (float*)d_out;

    const int* src = ei;
    const int* dst = ei + N_EDGES;

    // workspace layout (all offsets keep >=32B alignment)
    float* f = (float*)d_ws;
    const size_t NP = 50176;
    float* dinv = f;  f += NP;
    float* deg  = f;  f += NP;
    float* t    = f;  f += (size_t)N_NODES * HDIM;
    float* agg  = f;  f += (size_t)N_NODES * HDIM;
    float* hA   = f;  f += (size_t)N_NODES * HDIM;
    float* hB   = f;  f += (size_t)N_NODES * HDIM;
    float* logits = f; f += 64;
    _Float16* W2f = (_Float16*)f;              // 1 kstep * 4 tiles * 32 lanes * 16 halfs
    _Float16* W3f = W2f + 1 * 4 * 32 * 16;     // 2 ksteps

    auto cdiv = [](int a, int b) { return (a + b - 1) / b; };
    const int B = 256;

    // normalization: deg (with self-loop), dinv
    k_fill1<<<cdiv(N_NODES, B), B, 0, stream>>>(deg, N_NODES);
    k_deg  <<<cdiv(N_EDGES, B), B, 0, stream>>>(dst, deg, N_EDGES);
    k_dinv <<<cdiv(N_NODES, B), B, 0, stream>>>(deg, dinv, N_NODES);

    // ---- layer 1: 1 -> 10 ----
    k_l1        <<<cdiv(N_NODES * NCLS, B), B, 0, stream>>>(x, W1, t, N_NODES);
    k_zero      <<<cdiv(N_NODES * NCLS, B), B, 0, stream>>>(agg, N_NODES * NCLS);
    k_scatter10 <<<cdiv(N_EDGES, B), B, 0, stream>>>(t, src, dst, dinv, agg, N_EDGES);
    k_finish10p <<<cdiv(N_NODES * 16, B), B, 0, stream>>>(agg, t, dinv, b1, hA, N_NODES);

    // ---- layer 2: 10 -> 64 (WMMA, h1 padded to K-stride 16) ----
    k_wfrag<<<cdiv(1 * 4 * 32 * 16, B), B, 0, stream>>>(W2, W2f, 10, 1);
    k_gemm<1, 16><<<N_NODES / 16, 128, 0, stream>>>(hA, W2f, t);
    k_zero     <<<cdiv(N_NODES * HDIM, B), B, 0, stream>>>(agg, N_NODES * HDIM);
    k_scatter64<<<cdiv(N_EDGES * 16, B), B, 0, stream>>>((const float4*)t, src, dst, dinv, agg, N_EDGES);
    k_finish64 <<<cdiv(N_NODES * 16, B), B, 0, stream>>>((const float4*)agg, (const float4*)t,
                                                         dinv, (const float4*)b2, (float4*)hB, N_NODES);

    // ---- layer 3: 64 -> 64 (WMMA, 2 k-steps) ----
    k_wfrag<<<cdiv(2 * 4 * 32 * 16, B), B, 0, stream>>>(W3, W3f, 64, 2);
    k_gemm<2, 64><<<N_NODES / 16, 128, 0, stream>>>(hB, W3f, t);
    k_zero     <<<cdiv(N_NODES * HDIM, B), B, 0, stream>>>(agg, N_NODES * HDIM);
    k_scatter64<<<cdiv(N_EDGES * 16, B), B, 0, stream>>>((const float4*)t, src, dst, dinv, agg, N_EDGES);
    k_finish64 <<<cdiv(N_NODES * 16, B), B, 0, stream>>>((const float4*)agg, (const float4*)t,
                                                         dinv, (const float4*)b3, (float4*)hA, N_NODES);

    // ---- fc + softmax ----
    k_zero   <<<1, 64, 0, stream>>>(logits, 64);
    k_fc     <<<1024, 256, 0, stream>>>(hA, fcW, logits, N_NODES * HDIM / 2);
    k_softmax<<<1, 32, 0, stream>>>(logits, fcb, out);
}
